// HierarchicalSparsemaxAttentionNetwork_13134009991849
// MI455X (gfx1250) — matrix operand order repaired and checked by reading, not compiled, not run on hardware
//
#include <hip/hip_runtime.h>

// MI455X / gfx1250: bf16 WMMA for all large GEMMs, fp32 recurrent state,
// bf16 h-state storage, fused attention, double-buffered LDS staging,
// branchless K-tail handling, uniform scalar guards, global prefetch.

typedef __attribute__((ext_vector_type(16))) __bf16 v16bf;
typedef __attribute__((ext_vector_type(8)))  float  v8f;

#define G3 768    // 3*H
#define HH 256    // H
#define DD 512    // 2*H

// Load a 16-element bf16 fragment from two 16-byte-aligned 8-element runs.
static __device__ __forceinline__ v16bf ld_frag(const __bf16* p0, const __bf16* p1) {
  union { uint4 u[2]; v16bf v; } t;
  t.u[0] = *(const uint4*)p0;
  t.u[1] = *(const uint4*)p1;
  return t.v;
}

// ---------------------------------------------------------------------------
// Fused GRU gate GEMM.  blockIdx.z selects (dir, src):
//   src==0: Out=Gx[dir] = x_t @ Wx_dir + bx_dir   (x gathered via tokens, K=xK)
//   src==1: Out=Gh[dir] = h_{t-1} @ Wh_dir + bh_dir (K=256; zeros if first)
// Block = 128 threads (4 waves), tile = 32(M) x 64(N), K-chunks of 32,
// double-buffered LDS, 2 WMMAs per wave per chunk (B-fragment reuse).
// ---------------------------------------------------------------------------
__global__ __launch_bounds__(128) void gru_gate_gemm(
    const int* __restrict__ tok, const float* __restrict__ xsrc,
    const float* __restrict__ Wx_f, const float* __restrict__ Wh_f,
    const float* __restrict__ bx_f, const float* __restrict__ bh_f,
    const float* __restrict__ Wx_b, const float* __restrict__ Wh_b,
    const float* __restrict__ bx_b, const float* __restrict__ bh_b,
    const float* __restrict__ hprev, float* __restrict__ Gx, float* __restrict__ Gh,
    int M, int T, int xK, int step, int first)
{
  __shared__ __bf16 lA[2][32 * 40];   // [buf][row][k], stride 40
  __shared__ __bf16 lB[2][64 * 40];   // [buf][col][k] (transposed)

  const int tid = threadIdx.x;
  const int dir = blockIdx.z >> 1;
  const int src = blockIdx.z & 1;
  const int t   = dir ? (T - 1 - step) : step;

  const float* W;  const float* bias;  float* Out;  int K;
  if (src == 0) { K = xK; W = dir ? Wx_b : Wx_f; bias = dir ? bx_b : bx_f; Out = Gx; }
  else          { K = HH; W = dir ? Wh_b : Wh_f; bias = dir ? bh_b : bh_f; Out = Gh; }
  Out += (size_t)dir * M * G3;

  // Uniform (block-scalar) guard: A contribution is zero on the first
  // recurrent step (h_{-1} == 0).
  const bool skipA = (src == 1) && (first != 0);

  const int mtile = blockIdx.y * 32;
  const int nbase = blockIdx.x * 64;

  // Per-thread A-row pointer (row = tid>>2, one row per 4 threads).
  const float* arow;
  {
    int m = mtile + (tid >> 2);
    if (src == 0) {
      if (tok) arow = xsrc + (size_t)tok[m * T + t] * xK;
      else     arow = xsrc + ((size_t)m * T + t) * xK;
    } else {
      arow = hprev + ((size_t)dir * M + m) * HH;
    }
  }

  const int KB = (K + 31) >> 5;

  // stage A: 32 rows x 32 k, 8 bf16 per thread, one b128 LDS store.
  auto stageA = [&](int kb, int b) {
    const int r  = tid >> 2;
    const int k0 = (tid & 3) << 3;
    uint4 pk = {0u, 0u, 0u, 0u};
    if (!skipA) {                                     // uniform scalar branch
      const int gk  = (kb << 5) + k0;
      const int gkc = min(gk, K - 8);                 // K always multiple of 8
      union { float4 f4[2]; float f[8]; } ld;
      ld.f4[0] = *(const float4*)(arow + gkc);
      ld.f4[1] = *(const float4*)(arow + gkc + 4);
      union { uint4 u; __bf16 h[8]; } o;
      if ((kb << 5) + 32 <= K) {                      // uniform fast path
#pragma unroll
        for (int j = 0; j < 8; ++j) o.h[j] = (__bf16)ld.f[j];
      } else {                                        // branchless tail
#pragma unroll
        for (int j = 0; j < 8; ++j) o.h[j] = (__bf16)((gk + j < K) ? ld.f[j] : 0.f);
      }
      pk = o.u;
    }
    *(uint4*)&lA[b][r * 40 + k0] = pk;
  };

  // stage B: 32 k x 64 cols transposed; thread does 2 k-rows x 8 cols,
  // paired bf16 written as b32 stores.  Prefetch two chunks ahead
  // (speculative global_prefetch_b8; dropped silently if OOB).
  auto stageB = [&](int kb, int b) {
    const int kr2 = (tid >> 3) << 1;                  // 0,2,..,30
    const int c0  = (tid & 7) << 3;                   // 0..56
    const int gk0 = (kb << 5) + kr2, gk1 = gk0 + 1;
    union { float4 f4[2]; float f[8]; } r0, r1;
    const float* p0 = W + (size_t)min(gk0, K - 1) * G3 + nbase + c0;
    const float* p1 = W + (size_t)min(gk1, K - 1) * G3 + nbase + c0;
    r0.f4[0] = *(const float4*)p0;  r0.f4[1] = *(const float4*)(p0 + 4);
    r1.f4[0] = *(const float4*)p1;  r1.f4[1] = *(const float4*)(p1 + 4);
    if (gk0 + 64 < K) __builtin_prefetch(p0 + (size_t)64 * G3, 0, 1);
    const bool m0 = gk0 < K, m1 = gk1 < K;
#pragma unroll
    for (int c = 0; c < 8; ++c) {
      union { unsigned u; __bf16 h[2]; } p;
      p.h[0] = (__bf16)(m0 ? r0.f[c] : 0.f);
      p.h[1] = (__bf16)(m1 ? r1.f[c] : 0.f);
      *(unsigned*)&lB[b][(c0 + c) * 40 + kr2] = p.u;
    }
  };

  const int lane = tid & 31, wv = tid >> 5;
  const int hi = lane >> 4, l15 = lane & 15;

  v8f acc0 = {0.f, 0.f, 0.f, 0.f, 0.f, 0.f, 0.f, 0.f};
  v8f acc1 = {0.f, 0.f, 0.f, 0.f, 0.f, 0.f, 0.f, 0.f};

  stageA(0, 0); stageB(0, 0);
  __syncthreads();
  int buf = 0;
  for (int kb = 0; kb < KB; ++kb) {
    if (kb + 1 < KB) { stageA(kb + 1, buf ^ 1); stageB(kb + 1, buf ^ 1); }
    const __bf16* bp  = &lB[buf][(wv * 16 + l15) * 40 + hi * 8];
    v16bf bf = ld_frag(bp, bp + 16);
    const __bf16* a0p = &lA[buf][l15 * 40 + hi * 8];
    v16bf a0 = ld_frag(a0p, a0p + 16);
    const __bf16* a1p = &lA[buf][(16 + l15) * 40 + hi * 8];
    v16bf a1 = ld_frag(a1p, a1p + 16);
    acc0 = __builtin_amdgcn_wmma_f32_16x16x32_bf16(false, a0, false, bf,
                                                   (short)0, acc0, false, false);
    acc1 = __builtin_amdgcn_wmma_f32_16x16x32_bf16(false, a1, false, bf,
                                                   (short)0, acc1, false, false);
    __syncthreads();
    buf ^= 1;
  }

  const int col = nbase + wv * 16 + l15;
  const float bv = bias[col];
#pragma unroll
  for (int r = 0; r < 8; ++r) {
    Out[(size_t)(mtile + r + hi * 8) * G3 + col]      = acc0[r] + bv;
    Out[(size_t)(mtile + 16 + r + hi * 8) * G3 + col] = acc1[r] + bv;
  }
}

// ---------------------------------------------------------------------------
// GRU elementwise update: r/z/n gates, writes fp32 state + bf16 h history.
// ---------------------------------------------------------------------------
__global__ __launch_bounds__(256) void gru_update(
    const float* __restrict__ Gx, const float* __restrict__ Gh,
    float* __restrict__ hprev, __bf16* __restrict__ hstore,
    int M, int T, int step, int first)
{
  int idx = blockIdx.x * 256 + threadIdx.x;
  int tot = 2 * M * HH;
  if (idx >= tot) return;
  int dir = idx / (M * HH);
  int rem = idx - dir * M * HH;
  int n = rem >> 8, j = rem & 255;

  size_t gb = ((size_t)dir * M + n) * G3;
  float xr = Gx[gb + j], xz = Gx[gb + HH + j], xn = Gx[gb + 2 * HH + j];
  float hr = Gh[gb + j], hz = Gh[gb + HH + j], hn = Gh[gb + 2 * HH + j];
  size_t hb = ((size_t)dir * M + n) * HH + j;
  float h = first ? 0.f : hprev[hb];
  float r = 1.f / (1.f + __expf(-(xr + hr)));
  float z = 1.f / (1.f + __expf(-(xz + hz)));
  float nn = tanhf(xn + r * hn);
  float hnew = (1.f - z) * nn + z * h;
  hprev[hb] = hnew;
  int t = dir ? (T - 1 - step) : step;
  hstore[(((size_t)n * T) + t) * DD + dir * HH + j] = (__bf16)hnew;
}

// ---------------------------------------------------------------------------
// Fused attention score: u[m] = sum_c tanh( (H[m,:] @ W)[c] + b[c] ) * ctx[c]
// H bf16 [Mrows][512], W fp32 [512][512].  Block = 32 rows, 4 waves,
// double-buffered B staging, 2 WMMAs per chunk per wave.
// ---------------------------------------------------------------------------
__global__ __launch_bounds__(128) void attn_u_kernel(
    const __bf16* __restrict__ Hm, const float* __restrict__ W,
    const float* __restrict__ bias, const float* __restrict__ ctx,
    float* __restrict__ u)
{
  __shared__ __bf16 lA[32 * 520];    // full 32x512 bf16 A tile
  __shared__ __bf16 lB[2][64 * 40];
  __shared__ float usum[32];

  const int tid = threadIdx.x;
  const int mbase = blockIdx.x * 32;
  if (tid < 32) usum[tid] = 0.f;

  // stage A fully: 128 bf16 per thread as 16x b128
  {
    int r  = tid >> 2;          // 0..31
    int k0 = (tid & 3) * 128;   // 0,128,256,384
    const uint4* s = (const uint4*)(Hm + ((size_t)(mbase + r)) * DD + k0);
    uint4* d = (uint4*)&lA[r * 520 + k0];
#pragma unroll
    for (int q = 0; q < 16; ++q) d[q] = s[q];
  }
  __syncthreads();

  auto stageB = [&](int kb, int b, int nbase) {
    const int kr2 = (tid >> 3) << 1;
    const int c0  = (tid & 7) << 3;
    const int gk0 = (kb << 5) + kr2;
    union { float4 f4[2]; float f[8]; } r0, r1;
    const float* p0 = W + (size_t)gk0 * DD + nbase + c0;
    const float* p1 = p0 + DD;
    r0.f4[0] = *(const float4*)p0;  r0.f4[1] = *(const float4*)(p0 + 4);
    r1.f4[0] = *(const float4*)p1;  r1.f4[1] = *(const float4*)(p1 + 4);
    if (gk0 + 64 < DD) __builtin_prefetch(p0 + (size_t)64 * DD, 0, 1);
#pragma unroll
    for (int c = 0; c < 8; ++c) {
      union { unsigned u; __bf16 h[2]; } p;
      p.h[0] = (__bf16)r0.f[c];
      p.h[1] = (__bf16)r1.f[c];
      *(unsigned*)&lB[b][(c0 + c) * 40 + kr2] = p.u;
    }
  };

  const int lane = tid & 31, wv = tid >> 5;
  const int hi = lane >> 4, l15 = lane & 15;

  for (int i = 0; i < 8; ++i) {          // 8 groups of 64 output columns
    const int nbase = i * 64;
    v8f acc0 = {0.f, 0.f, 0.f, 0.f, 0.f, 0.f, 0.f, 0.f};
    v8f acc1 = {0.f, 0.f, 0.f, 0.f, 0.f, 0.f, 0.f, 0.f};
    stageB(0, 0, nbase);
    __syncthreads();
    int buf = 0;
    for (int kb = 0; kb < 16; ++kb) {    // K = 512
      if (kb + 1 < 16) stageB(kb + 1, buf ^ 1, nbase);
      const __bf16* bp  = &lB[buf][(wv * 16 + l15) * 40 + hi * 8];
      v16bf bf = ld_frag(bp, bp + 16);
      const __bf16* a0p = &lA[l15 * 520 + (kb << 5) + hi * 8];
      v16bf a0 = ld_frag(a0p, a0p + 16);
      const __bf16* a1p = &lA[(16 + l15) * 520 + (kb << 5) + hi * 8];
      v16bf a1 = ld_frag(a1p, a1p + 16);
      acc0 = __builtin_amdgcn_wmma_f32_16x16x32_bf16(false, a0, false, bf,
                                                     (short)0, acc0, false, false);
      acc1 = __builtin_amdgcn_wmma_f32_16x16x32_bf16(false, a1, false, bf,
                                                     (short)0, acc1, false, false);
      __syncthreads();
      buf ^= 1;
    }
    const int col = nbase + wv * 16 + l15;
    const float cv = ctx[col], bv = bias[col];
#pragma unroll
    for (int r = 0; r < 8; ++r) {
      atomicAdd(&usum[r + hi * 8],      tanhf(acc0[r] + bv) * cv);
      atomicAdd(&usum[16 + r + hi * 8], tanhf(acc1[r] + bv) * cv);
    }
  }
  __syncthreads();
  if (tid < 32) u[mbase + tid] = usum[tid];
}

// ---------------------------------------------------------------------------
// Exact sparsemax along rows of length T (power of two <= 64), in place.
// ---------------------------------------------------------------------------
__global__ void sparsemax_kernel(float* __restrict__ u, int T)
{
  __shared__ float ss[64];
  __shared__ float sx[64];
  __shared__ float sh[2];
  const int row = blockIdx.x, tid = threadIdx.x;
  float x = u[(size_t)row * T + tid];
  sx[tid] = x;
  __syncthreads();
  if (tid == 0) {
    float m = sx[0];
    for (int i = 1; i < T; ++i) m = fmaxf(m, sx[i]);
    sh[0] = m;
  }
  __syncthreads();
  x -= sh[0];
  ss[tid] = x;
  __syncthreads();
  // bitonic sort, descending
  for (int k = 2; k <= T; k <<= 1)
    for (int j = k >> 1; j > 0; j >>= 1) {
      int ixj = tid ^ j;
      if (ixj > tid) {
        float a = ss[tid], b = ss[ixj];
        bool desc = ((tid & k) == 0);
        if (desc ? (a < b) : (a > b)) { ss[tid] = b; ss[ixj] = a; }
      }
      __syncthreads();
    }
  if (tid == 0) {
    float cs = 0.f, csAt = ss[0];
    int supp = 1;
    for (int i = 0; i < T; ++i) {
      cs += ss[i];
      if ((float)(i + 1) * ss[i] > cs - 1.0f) { supp = i + 1; csAt = cs; }
    }
    sh[1] = (csAt - 1.0f) / (float)supp;
  }
  __syncthreads();
  u[(size_t)row * T + tid] = fmaxf(x - sh[1], 0.f);
}

// ---------------------------------------------------------------------------
// out[n][d] = sum_l att[n][l] * H[n][l][d]   (H bf16, fp32 accumulate)
// grid.x = rows*2, block 256 (d split across two blocks per row)
// ---------------------------------------------------------------------------
__global__ __launch_bounds__(256) void attn_wsum_kernel(
    const float* __restrict__ att, const __bf16* __restrict__ Hm,
    float* __restrict__ out, int T)
{
  __shared__ float aw[64];
  const int n = blockIdx.x >> 1;
  const int d = ((blockIdx.x & 1) << 8) + threadIdx.x;
  if (threadIdx.x < T) aw[threadIdx.x] = att[(size_t)n * T + threadIdx.x];
  __syncthreads();
  float s = 0.f;
  for (int l = 0; l < T; ++l)
    s += aw[l] * (float)Hm[((size_t)n * T + l) * DD + d];
  out[(size_t)n * DD + d] = s;
}

// ---------------------------------------------------------------------------
// Final projection: out[32][10] = doc[32][512] @ W[512][10] + b
// ---------------------------------------------------------------------------
__global__ void out_proj_kernel(const float* __restrict__ doc,
                                const float* __restrict__ W,
                                const float* __restrict__ b,
                                float* __restrict__ out)
{
  int idx = blockIdx.x * blockDim.x + threadIdx.x;
  if (idx >= 320) return;
  int bi = idx / 10, c = idx % 10;
  float s = b[c];
  for (int d = 0; d < DD; ++d) s += doc[(size_t)bi * DD + d] * W[d * 10 + c];
  out[idx] = s;
}

// ---------------------------------------------------------------------------
extern "C" void kernel_launch(void* const* d_in, const int* in_sizes, int n_in,
                              void* d_out, int out_size, void* d_ws, size_t ws_size,
                              hipStream_t stream) {
  (void)in_sizes; (void)n_in; (void)out_size; (void)ws_size;
  const int*   tokens = (const int*)d_in[0];
  const float* emb    = (const float*)d_in[1];
  const float *wWxf = (const float*)d_in[2],  *wWhf = (const float*)d_in[3];
  const float *wbxf = (const float*)d_in[4],  *wbhf = (const float*)d_in[5];
  const float *wWxb = (const float*)d_in[6],  *wWhb = (const float*)d_in[7];
  const float *wbxb = (const float*)d_in[8],  *wbhb = (const float*)d_in[9];
  const float *wLinW = (const float*)d_in[10], *wLinB = (const float*)d_in[11];
  const float *wCtx  = (const float*)d_in[12];
  const float *sWxf = (const float*)d_in[13], *sWhf = (const float*)d_in[14];
  const float *sbxf = (const float*)d_in[15], *sbhf = (const float*)d_in[16];
  const float *sWxb = (const float*)d_in[17], *sWhb = (const float*)d_in[18];
  const float *sbxb = (const float*)d_in[19], *sbhb = (const float*)d_in[20];
  const float *sLinW = (const float*)d_in[21], *sLinB = (const float*)d_in[22];
  const float *sCtx  = (const float*)d_in[23];
  const float *outW = (const float*)d_in[24], *outB = (const float*)d_in[25];
  float* out = (float*)d_out;

  char* ws = (char*)d_ws;
  size_t off = 0;
  auto take = [&](size_t bytes) -> char* {
    char* p = ws + off;
    off = (off + bytes + 255) & ~(size_t)255;
    return p;
  };
  __bf16* h_word  = (__bf16*)take((size_t)1024 * 64 * DD * 2);   // 64 MiB
  __bf16* h_sent  = (__bf16*)take((size_t)32 * 32 * DD * 2);
  float*  hprev_w = (float*)take((size_t)2 * 1024 * HH * 4);
  float*  hprev_s = (float*)take((size_t)2 * 32 * HH * 4);
  float*  Gx_w    = (float*)take((size_t)2 * 1024 * G3 * 4);
  float*  Gh_w    = (float*)take((size_t)2 * 1024 * G3 * 4);
  float*  Gx_s    = (float*)take((size_t)2 * 32 * G3 * 4);
  float*  Gh_s    = (float*)take((size_t)2 * 32 * G3 * 4);
  float*  u_w     = (float*)take((size_t)1024 * 64 * 4);
  float*  u_s     = (float*)take((size_t)32 * 32 * 4);
  float*  sents   = (float*)take((size_t)32 * 32 * DD * 4);
  float*  doc     = (float*)take((size_t)32 * DD * 4);

  // ---- word-level BiGRU (1024 sequences x 64 steps)
  for (int step = 0; step < 64; ++step) {
    gru_gate_gemm<<<dim3(12, 32, 4), 128, 0, stream>>>(
        tokens, emb, wWxf, wWhf, wbxf, wbhf, wWxb, wWhb, wbxb, wbhb,
        hprev_w, Gx_w, Gh_w, 1024, 64, 200, step, step == 0);
    gru_update<<<2048, 256, 0, stream>>>(Gx_w, Gh_w, hprev_w, h_word,
                                         1024, 64, step, step == 0);
  }
  // ---- word attention + sentence vectors
  attn_u_kernel<<<2048, 128, 0, stream>>>(h_word, wLinW, wLinB, wCtx, u_w);
  sparsemax_kernel<<<1024, 64, 0, stream>>>(u_w, 64);
  attn_wsum_kernel<<<2048, 256, 0, stream>>>(u_w, h_word, sents, 64);

  // ---- sentence-level BiGRU (32 sequences x 32 steps)
  for (int step = 0; step < 32; ++step) {
    gru_gate_gemm<<<dim3(12, 1, 4), 128, 0, stream>>>(
        nullptr, sents, sWxf, sWhf, sbxf, sbhf, sWxb, sWhb, sbxb, sbhb,
        hprev_s, Gx_s, Gh_s, 32, 32, 512, step, step == 0);
    gru_update<<<64, 256, 0, stream>>>(Gx_s, Gh_s, hprev_s, h_sent,
                                       32, 32, step, step == 0);
  }
  // ---- sentence attention + doc vector + output
  attn_u_kernel<<<32, 128, 0, stream>>>(h_sent, sLinW, sLinB, sCtx, u_s);
  sparsemax_kernel<<<32, 32, 0, stream>>>(u_s, 32);
  attn_wsum_kernel<<<64, 256, 0, stream>>>(u_s, h_sent, doc, 32);
  out_proj_kernel<<<2, 160, 0, stream>>>(doc, outW, outB, out);
}
